// E74v2Cell_8907762172393
// MI455X (gfx1250) — matrix operand order repaired and checked by reading, not compiled, not run on hardware
//
#include <hip/hip_runtime.h>
#include <cmath>

// ---------------- problem constants (from reference) ----------------
#define T_DIM 2048
#define B_DIM 16
#define D_DIM 1024
#define N_DIM 64
#define MROWS (T_DIM * B_DIM)          // 32768 flattened (t,b) rows
#define TBN   ((size_t)MROWS * N_DIM)  // elements per projection array

// ---------------- CDNA5 WMMA types ----------------
typedef __attribute__((ext_vector_type(16))) __bf16 v16bf;
typedef __attribute__((ext_vector_type(16))) short  v16s;
typedef __attribute__((ext_vector_type(8)))  short  v8s;
typedef __attribute__((ext_vector_type(8)))  float  v8f;

union FragU { v16s s; v16bf b; };

__device__ __forceinline__ unsigned short f32_bf16_rne(float f) {
  unsigned int u = __float_as_uint(f);
  u += 0x7FFFu + ((u >> 16) & 1u);          // round-to-nearest-even
  return (unsigned short)(u >> 16);
}

// LDS row strides in ushorts: 40 ushorts = 80 B = 5*16 B -> b128-aligned rows,
// and 20*lane mod 64 covers all multiples of 4 -> conflict-free b128 reads.
#define ASTR 40
#define BSTR 40

// =====================================================================
// Kernel 1: fused projection GEMM, WMMA bf16 path.
//   Block tile: M=32 rows of x, N=320 channels (all 5 weight matrices).
//   8 waves: (wave>>2) picks M half-tile, (wave&3)+4*rep picks N tile.
//   K loop: 32 per iteration, 32 iterations over D=1024.
//   Epilogue fuses bias + sigmoid for e (rep==3) and w (rep==4).
// =====================================================================
__global__ __launch_bounds__(256) void proj_gemm_wmma(
    const float* __restrict__ x,
    const float* __restrict__ Wk, const float* __restrict__ Wv,
    const float* __restrict__ Wq, const float* __restrict__ We,
    const float* __restrict__ Ww,
    const float* __restrict__ be, const float* __restrict__ bw,
    float* __restrict__ proj)
{
  __shared__ __align__(16) unsigned short Abuf[32  * ASTR];   // 32x32 bf16 tile of x
  __shared__ __align__(16) unsigned short Bbuf[320 * BSTR];   // 320x32 bf16 tile of W's

  const int tid  = threadIdx.x;
  const int wave = tid >> 5;        // wave32
  const int lane = tid & 31;
  const int half = lane >> 4;       // 0: lanes 0-15, 1: lanes 16-31
  const int lr   = lane & 15;
  const int mBase = blockIdx.x * 32;

  v8f acc[5];
  #pragma unroll
  for (int i = 0; i < 5; ++i)
    acc[i] = (v8f){0.f, 0.f, 0.f, 0.f, 0.f, 0.f, 0.f, 0.f};

  const int ar = tid >> 3;          // staging row 0..31
  const int kq = (tid & 7) * 4;     // staging k offset 0,4,...,28

  for (int kk = 0; kk < D_DIM; kk += 32) {
    // ---- stage A tile (32 rows x 32 K), f32 -> bf16 ----
    {
      const float4 v = *(const float4*)(x + (size_t)(mBase + ar) * D_DIM + kk + kq);
      unsigned short* dst = &Abuf[ar * ASTR + kq];
      dst[0] = f32_bf16_rne(v.x); dst[1] = f32_bf16_rne(v.y);
      dst[2] = f32_bf16_rne(v.z); dst[3] = f32_bf16_rne(v.w);
    }
    // ---- stage B tile: 5 matrices x 64 rows x 32 K ----
    #pragma unroll
    for (int mat = 0; mat < 5; ++mat) {
      const float* W = (mat == 0) ? Wk : (mat == 1) ? Wv : (mat == 2) ? Wq
                       : (mat == 3) ? We : Ww;
      #pragma unroll
      for (int rep = 0; rep < 2; ++rep) {
        const int n = rep * 32 + ar;                       // 0..63
        const float4 v = *(const float4*)(W + (size_t)n * D_DIM + kk + kq);
        unsigned short* dst = &Bbuf[(mat * 64 + n) * BSTR + kq];
        dst[0] = f32_bf16_rne(v.x); dst[1] = f32_bf16_rne(v.y);
        dst[2] = f32_bf16_rne(v.z); dst[3] = f32_bf16_rne(v.w);
      }
    }
    __syncthreads();

    // ---- A fragment: 16-bit A layout (ISA 7.12.2):
    //   lane<16: K{0..7,16..23}; lane>=16: K{8..15,24..31}, row = M = lr ----
    FragU fa;
    {
      const int row = (wave >> 2) * 16 + lr;
      const v8s lo = *(const v8s*)&Abuf[row * ASTR + half * 8];
      const v8s hi = *(const v8s*)&Abuf[row * ASTR + 16 + half * 8];
      #pragma unroll
      for (int i = 0; i < 8; ++i) { fa.s[i] = lo[i]; fa.s[8 + i] = hi[i]; }
    }

    // ---- 5 N-tiles per wave ----
    #pragma unroll
    for (int rep = 0; rep < 5; ++rep) {
      const int nt = (wave & 3) + 4 * rep;    // 0..19
      const int c  = nt * 16 + lr;            // channel 0..319 (== B column)
      FragU fb;                                // B: col = lane%16, K{0..7,16..23} / K{8..15,24..31} by half
      const v8s blo = *(const v8s*)&Bbuf[c * BSTR + half * 8];
      const v8s bhi = *(const v8s*)&Bbuf[c * BSTR + 16 + half * 8];
      #pragma unroll
      for (int i = 0; i < 8; ++i) { fb.s[i] = blo[i]; fb.s[8 + i] = bhi[i]; }

      acc[rep] = __builtin_amdgcn_wmma_f32_16x16x32_bf16(
          false, fa.b, false, fb.b, (short)0, acc[rep], false, false);
    }
    __syncthreads();
  }

  // ---- epilogue: C layout -> proj[rep][m][ch]; sigmoid+bias on e,w ----
  const int ch = (wave & 3) * 16 + lr;   // channel within its 64-wide array
  #pragma unroll
  for (int rep = 0; rep < 5; ++rep) {
    #pragma unroll
    for (int r = 0; r < 8; ++r) {
      const int m = mBase + (wave >> 2) * 16 + half * 8 + r;
      float v = acc[rep][r];
      if (rep == 3) v = 1.f / (1.f + __expf(-(v + be[ch])));
      if (rep == 4) v = 1.f / (1.f + __expf(-(v + bw[ch])));
      proj[(size_t)rep * TBN + (size_t)m * N_DIM + ch] = v;
    }
  }
}

// =====================================================================
// Kernel 2: sequential scan. One workgroup per batch; S (64x64) lives in
// registers (16 elems/thread). Per step: LDS broadcast of k,v,q,e,w;
// rank-1 state update S += k*(w*v - e*S); fused partial dot with q;
// 4-way LDS reduce; tanh; store. S_final written after the loop.
// =====================================================================
__global__ __launch_bounds__(256) void scan_kernel(
    const float* __restrict__ proj,  // 5 arrays, each MROWS x 64
    const float* __restrict__ S0,    // B x 64 x 64
    float* __restrict__ out,         // T x B x 64
    float* __restrict__ Sfin)        // B x 64 x 64
{
  __shared__ float kv[64], vv[64], qv[64], ev[64], wv_[64];
  __shared__ float part[256];

  const int tid = threadIdx.x;
  const int b   = blockIdx.x;
  const int i   = tid & 63;    // state row
  const int jb  = tid >> 6;    // column block 0..3 (16 cols each)

  float S[16];
  #pragma unroll
  for (int jj = 0; jj < 16; ++jj)
    S[jj] = S0[(size_t)b * 4096 + i * 64 + jb * 16 + jj];

  const float* kbuf = proj + 0 * TBN;
  const float* vbuf = proj + 1 * TBN;
  const float* qbuf = proj + 2 * TBN;
  const float* ebuf = proj + 3 * TBN;
  const float* wbuf = proj + 4 * TBN;

  for (int t = 0; t < T_DIM; ++t) {
    const size_t rowoff = ((size_t)t * B_DIM + b) * 64;
    if      (jb == 0) kv[i] = kbuf[rowoff + i];
    else if (jb == 1) vv[i] = vbuf[rowoff + i];
    else if (jb == 2) qv[i] = qbuf[rowoff + i];
    else              ev[i] = ebuf[rowoff + i];
    if (tid < 64)     wv_[i] = wbuf[rowoff + i];
    __syncthreads();

    const float ei  = ev[i];
    const float tmp = wv_[i] * vv[i];
    float partial = 0.f;
    #pragma unroll
    for (int jj = 0; jj < 16; ++jj) {
      const int j  = jb * 16 + jj;
      const float kj = kv[j];
      // S*(1-e*k) + (w*v)*k  ==  S + k*(tmp - e*S)
      S[jj] = fmaf(kj, fmaf(-ei, S[jj], tmp), S[jj]);
      partial = fmaf(S[jj], qv[j], partial);
    }
    part[tid] = partial;
    __syncthreads();
    if (tid < 64) {
      const float o = part[tid] + part[tid + 64] + part[tid + 128] + part[tid + 192];
      out[(size_t)t * (B_DIM * 64) + b * 64 + i] = tanhf(o);
    }
    // hazards: next-iter vector writes touch kv/.../wv_ (not part[]); the
    // barrier after those writes orders everything before part[] is reused.
  }

  #pragma unroll
  for (int jj = 0; jj < 16; ++jj)
    Sfin[(size_t)b * 4096 + i * 64 + jb * 16 + jj] = S[jj];
}

// =====================================================================
extern "C" void kernel_launch(void* const* d_in, const int* in_sizes, int n_in,
                              void* d_out, int out_size, void* d_ws, size_t ws_size,
                              hipStream_t stream) {
  const float* x  = (const float*)d_in[0];
  const float* S0 = (const float*)d_in[1];
  const float* Wk = (const float*)d_in[2];
  const float* Wv = (const float*)d_in[3];
  const float* Wq = (const float*)d_in[4];
  const float* We = (const float*)d_in[5];
  const float* be = (const float*)d_in[6];
  const float* Ww = (const float*)d_in[7];
  const float* bw = (const float*)d_in[8];

  float* out  = (float*)d_out;                        // output then S_final
  float* proj = (float*)d_ws;                         // 5 * 32768 * 64 f32 = 40 MB

  proj_gemm_wmma<<<dim3(MROWS / 32), 256, 0, stream>>>(
      x, Wk, Wv, Wq, We, Ww, be, bw, proj);

  scan_kernel<<<dim3(B_DIM), 256, 0, stream>>>(
      proj, S0, out, out + (size_t)T_DIM * B_DIM * N_DIM);
}
